// ParallelTransformerBlock_64811056497151
// MI455X (gfx1250) — compile-verified, matrix operand order checked by reading, not run on hardware
//
#include <hip/hip_runtime.h>
#include <hip/hip_bf16.h>
#include <stdint.h>

// ---------------------------------------------------------------------------
// CDNA5 (gfx1250) ParallelTransformerBlock, f16 WMMA pipeline, f32 accum.
// Async (memory -> LDS) staging via GLOBAL_LOAD_ASYNC_TO_LDS_B128.
// ---------------------------------------------------------------------------

typedef __attribute__((ext_vector_type(16))) _Float16 v16h;
typedef __attribute__((ext_vector_type(8)))  _Float16 v8h;
typedef __attribute__((ext_vector_type(8)))  float    v8f;

#define DIM        2048
#define HEADS      16
#define DIM_HEAD   128
#define FF_INNER   8192
#define ATTN_INNER 2048
#define FUSED_OUT  18688   // 2048 + 128 + 128 + 16384
#define SEQ        2048
#define BATCH      2
#define ROWS       (BATCH * SEQ)   // 4096

// Async global->LDS 16-byte copy (ASYNCcnt tracked).  The LDS address operand
// is the low 32 bits of the generic pointer.
// Fallback is a plain copy for the host compilation pass.
#if defined(__AMDGCN__)
#define ASYNC_LDS_B128(ldsp, gp)                                                \
    asm volatile("global_load_async_to_lds_b128 %0, %1, off"                    \
                 :: "v"((uint32_t)(uintptr_t)(ldsp)), "v"((const void*)(gp))    \
                 : "memory")
#define WAIT_ASYNC0() asm volatile("s_wait_asynccnt 0x0" ::: "memory")
#define WAIT_DS0()    asm volatile("s_wait_dscnt 0x0" ::: "memory")
#else
#define ASYNC_LDS_B128(ldsp, gp) (*(v8h*)(ldsp) = *(const v8h*)(gp))
#define WAIT_ASYNC0()
#define WAIT_DS0()
#endif

// Build a 16x32 f16 A/B-operand fragment for v_wmma_f32_16x16x32_f16 from an
// LDS row (row-contiguous in the K direction).  Per ISA 7.12.2:
// lanes 0-15 hold K {kh..kh+7, 16+kh..16+kh+7} with kh=0, lanes 16-31 kh=8.
__device__ __forceinline__ v16h make_frag(const _Float16* rowp, int kh) {
    v8h lo = *(const v8h*)(rowp + kh);
    v8h hi = *(const v8h*)(rowp + 16 + kh);
    return __builtin_shufflevector(lo, hi, 0,1,2,3,4,5,6,7,8,9,10,11,12,13,14,15);
}

__device__ __forceinline__ v8f wmma_f16(v16h a, v16h b, v8f c) {
    // D = A*B + C, 16x16x32, f32 accumulate
    return __builtin_amdgcn_wmma_f32_16x16x32_f16(false, a, false, b,
                                                  (short)0, c, false, false);
}

// ---------------------------------------------------------------------------
// LayerNorm (per 2048-wide row) + cast to f16
// ---------------------------------------------------------------------------
__global__ __launch_bounds__(256) void ln_cast_kernel(const float* __restrict__ x,
                                                      const float* __restrict__ gamma,
                                                      _Float16* __restrict__ xn) {
    const int row = blockIdx.x;
    const float* xr = x + (size_t)row * DIM;
    const int t = threadIdx.x;

    float v[8], s = 0.f, s2 = 0.f;
#pragma unroll
    for (int i = 0; i < 8; ++i) {
        v[i] = xr[i * 256 + t];
        s  += v[i];
        s2 += v[i] * v[i];
    }
#pragma unroll
    for (int m = 16; m >= 1; m >>= 1) {
        s  += __shfl_xor(s,  m, 32);
        s2 += __shfl_xor(s2, m, 32);
    }
    __shared__ float rs[8], rs2[8];
    const int w = t >> 5, lane = t & 31;
    if (lane == 0) { rs[w] = s; rs2[w] = s2; }
    __syncthreads();
    if (t == 0) {
        float a = 0.f, b2 = 0.f;
        for (int i = 0; i < 8; ++i) { a += rs[i]; b2 += rs2[i]; }
        rs[0] = a; rs2[0] = b2;
    }
    __syncthreads();
    const float mean = rs[0] * (1.0f / DIM);
    const float var  = rs2[0] * (1.0f / DIM) - mean * mean;
    const float rstd = rsqrtf(var + 1e-5f);
#pragma unroll
    for (int i = 0; i < 8; ++i) {
        const int idx = i * 256 + t;
        xn[(size_t)row * DIM + idx] = (_Float16)((v[i] - mean) * rstd * gamma[idx]);
    }
}

// ---------------------------------------------------------------------------
// f32 -> f16 cast (weights)
// ---------------------------------------------------------------------------
__global__ void cast_f32_f16_kernel(const float* __restrict__ in,
                                    _Float16* __restrict__ out, size_t n) {
    size_t i = (size_t)blockIdx.x * blockDim.x + threadIdx.x;
    const size_t stride = (size_t)gridDim.x * blockDim.x;
    for (; i < n; i += stride) out[i] = (_Float16)in[i];
}

// ---------------------------------------------------------------------------
// Tiled WMMA GEMM: C[M,N] = A[M,K] * B[K,N], f16 in, f32 accum.
// Block = 256 thr = 8 waves, tile 128x128, K step 32.
// Waves arranged 4x2; each wave owns a 32x64 sub-tile (2x4 wmma tiles).
// A tile staged with async global->LDS; B tile staged transposed manually.
// ---------------------------------------------------------------------------
template<bool OUT_F16, bool ACC>
__global__ __launch_bounds__(256) void gemm_wmma_kernel(const _Float16* __restrict__ A,
                                                        const _Float16* __restrict__ B,
                                                        float*    __restrict__ C32,
                                                        _Float16* __restrict__ C16,
                                                        int M, int N, int K) {
    constexpr int BM = 128, BN = 128, BK = 32;
    constexpr int AST = 40;   // padded row stride (halfwords); 80B rows, 16B aligned
    constexpr int BST = 40;
    __shared__ alignas(16) _Float16 As[BM * AST];          // [m][k]
    __shared__ alignas(16) _Float16 Bs[BN * BST];          // [n][k] (transposed)

    const int m0 = blockIdx.y * BM;
    const int n0 = blockIdx.x * BN;
    const int t = threadIdx.x;
    const int lane = t & 31, w = t >> 5;
    const int wm = (w >> 1) * 32;     // wave row offset in tile
    const int wn = (w & 1) * 64;      // wave col offset in tile
    const int hf = lane >> 4, l16 = lane & 15, kh = hf * 8;

    v8f acc[2][4] = {};

    for (int k0 = 0; k0 < K; k0 += BK) {
        // --- stage A tile (128x32): async global->LDS b128 ---
#pragma unroll
        for (int i = 0; i < 2; ++i) {
            const int c = t + i * 256;              // 512 chunks of 8 halfwords
            const int row = c >> 2, col = (c & 3) * 8;
            ASYNC_LDS_B128(&As[row * AST + col],
                           &A[(size_t)(m0 + row) * K + k0 + col]);
        }
        // --- stage B tile transposed into [n][k] ---
#pragma unroll
        for (int i = 0; i < 2; ++i) {
            const int c = t + i * 256;
            const int kk = c >> 4, nn = (c & 15) * 8;
            v8h tmp = *(const v8h*)&B[(size_t)(k0 + kk) * N + n0 + nn];
#pragma unroll
            for (int j = 0; j < 8; ++j) Bs[(nn + j) * BST + kk] = tmp[j];
        }
        if (k0 + BK < K) {   // global_prefetch_b8 for next B tile
            __builtin_prefetch(&B[(size_t)(k0 + BK + (t >> 4)) * N + n0 + ((t & 15) * 8)], 0, 1);
        }
        WAIT_ASYNC0();       // retire this wave's async tile before publishing
        __syncthreads();

        v16h af[2], bf[4];
#pragma unroll
        for (int i = 0; i < 2; ++i)
            af[i] = make_frag(&As[(wm + i * 16 + l16) * AST], kh);
#pragma unroll
        for (int j = 0; j < 4; ++j)
            bf[j] = make_frag(&Bs[(wn + j * 16 + l16) * BST], kh);
#pragma unroll
        for (int i = 0; i < 2; ++i)
#pragma unroll
            for (int j = 0; j < 4; ++j)
                acc[i][j] = wmma_f16(af[i], bf[j], acc[i][j]);
        __syncthreads();
    }

    // Epilogue: C tile layout per ISA (lane half selects rows 0-7 / 8-15)
#pragma unroll
    for (int i = 0; i < 2; ++i)
#pragma unroll
        for (int j = 0; j < 4; ++j)
#pragma unroll
            for (int r = 0; r < 8; ++r) {
                const int m  = m0 + wm + i * 16 + r + 8 * hf;
                const int nn = n0 + wn + j * 16 + l16;
                const float val = acc[i][j][r];
                if (OUT_F16) {
                    C16[(size_t)m * N + nn] = (_Float16)val;
                } else {
                    const size_t idx = (size_t)m * N + nn;
                    C32[idx] = ACC ? (C32[idx] + val) : val;
                }
            }
}

// ---------------------------------------------------------------------------
// RoPE + split fused -> q [b,h,n,d] (pre-scaled), k [b,n,d], v [b,n,d]
// ---------------------------------------------------------------------------
__global__ __launch_bounds__(256) void rope_split_kernel(const _Float16* __restrict__ fused,
                                                         _Float16* __restrict__ Qo,
                                                         _Float16* __restrict__ Ko,
                                                         _Float16* __restrict__ Vo) {
    const int tok = blockIdx.x;           // b*n tokens
    const int b = tok / SEQ, pos = tok % SEQ;
    const int t = threadIdx.x;
    __shared__ float cs[DIM_HEAD], sn[DIM_HEAD];
    if (t < DIM_HEAD) {
        const int j = t & 63;             // pos = concat(freqs, freqs)
        const float invf = __powf(10000.f, -(float)(2 * j) * (1.f / DIM_HEAD));
        const float ang = (float)pos * invf;
        cs[t] = __cosf(ang);
        sn[t] = __sinf(ang);
    }
    __syncthreads();
    const _Float16* fr = fused + (size_t)tok * FUSED_OUT;
    const float qscale = 0.08838834764831845f;   // 128^-0.5

    for (int e = t; e < ATTN_INNER; e += 256) {  // q
        const int hh = e >> 7, d = e & 127;
        const float v0 = (float)fr[e];
        const int dro = (d < 64) ? (d + 64) : (d - 64);
        const float vr = (float)fr[(e & ~127) + dro];
        const float o = v0 * cs[d] + ((d < 64) ? -vr : vr) * sn[d];
        Qo[(((size_t)(b * HEADS + hh) * SEQ) + pos) * DIM_HEAD + d] = (_Float16)(o * qscale);
    }
    if (t < DIM_HEAD) {                          // k (single head)
        const int d = t;
        const float v0 = (float)fr[ATTN_INNER + d];
        const int dro = (d < 64) ? (d + 64) : (d - 64);
        const float vr = (float)fr[ATTN_INNER + dro];
        const float o = v0 * cs[d] + ((d < 64) ? -vr : vr) * sn[d];
        Ko[((size_t)b * SEQ + pos) * DIM_HEAD + d] = (_Float16)o;
        Vo[((size_t)b * SEQ + pos) * DIM_HEAD + d] = fr[ATTN_INNER + DIM_HEAD + d];
    }
}

// ---------------------------------------------------------------------------
// Flash attention, WMMA.  Block: (b, h, 128 q-rows); 8 waves x 16 rows each.
// Streams 32-key tiles; online softmax with 16-lane shfl reductions.
// Q and K tiles staged with async global->LDS; V staged transposed manually.
// ---------------------------------------------------------------------------
__global__ __launch_bounds__(256) void attn_wmma_kernel(const _Float16* __restrict__ Q,
                                                        const _Float16* __restrict__ Km,
                                                        const _Float16* __restrict__ Vm,
                                                        const int* __restrict__ vl,
                                                        _Float16* __restrict__ O) {
    constexpr int QSTR = 136, KSTR = 136, VSTR = 40, PSTR = 40;
    __shared__ alignas(16) _Float16 Qs[128 * QSTR];      // [row][d]
    __shared__ alignas(16) _Float16 Ks[32 * KSTR];       // [key][d]
    __shared__ alignas(16) _Float16 Vs[128 * VSTR];      // [d][key]  (transposed)
    __shared__ alignas(16) _Float16 Ps[8 * 16 * PSTR];   // per-wave 16x32 P scratch

    const int nb0 = blockIdx.x * 128;
    const int h = blockIdx.y, b = blockIdx.z;
    const int t = threadIdx.x, lane = t & 31, w = t >> 5;
    const int hf = lane >> 4, l16 = lane & 15, kh = hf * 8;
    const int valid = vl[b];

    // stage Q tile 128x128 (async)
    const _Float16* qg = Q + ((size_t)(b * HEADS + h) * SEQ + nb0) * DIM_HEAD;
#pragma unroll
    for (int i = 0; i < 8; ++i) {
        const int c = t + i * 256;
        const int row = c >> 4, d0 = (c & 15) * 8;
        ASYNC_LDS_B128(&Qs[row * QSTR + d0], &qg[(size_t)row * DIM_HEAD + d0]);
    }
    WAIT_ASYNC0();
    __syncthreads();

    v16h qf[4];
#pragma unroll
    for (int kk = 0; kk < 4; ++kk)
        qf[kk] = make_frag(&Qs[(w * 16 + l16) * QSTR + kk * 32], kh);

    v8f acc[8] = {};
    float mrun[8], lrun[8];
#pragma unroll
    for (int r = 0; r < 8; ++r) { mrun[r] = -1e30f; lrun[r] = 0.f; }

    const _Float16* kg = Km + (size_t)b * SEQ * DIM_HEAD;
    const _Float16* vg = Vm + (size_t)b * SEQ * DIM_HEAD;

    for (int kt = 0; kt < SEQ; kt += 32) {
        __syncthreads();
        // stage K (32x128, async) and V transposed (128d x 32key, manual)
#pragma unroll
        for (int i = 0; i < 2; ++i) {
            const int c = t + i * 256;
            const int key = c >> 4, d0 = (c & 15) * 8;
            ASYNC_LDS_B128(&Ks[key * KSTR + d0],
                           &kg[(size_t)(kt + key) * DIM_HEAD + d0]);
            v8h tv = *(const v8h*)&vg[(size_t)(kt + key) * DIM_HEAD + d0];
#pragma unroll
            for (int j = 0; j < 8; ++j) Vs[(d0 + j) * VSTR + key] = tv[j];
        }
        WAIT_ASYNC0();
        __syncthreads();

        // sim: q(16x128) . k^T(128x32)  -> two 16x16 frags
        v8f s0 = {}, s1 = {};
#pragma unroll
        for (int kk = 0; kk < 4; ++kk) {
            v16h b0 = make_frag(&Ks[(l16) * KSTR + kk * 32], kh);
            v16h b1 = make_frag(&Ks[(16 + l16) * KSTR + kk * 32], kh);
            s0 = wmma_f16(qf[kk], b0, s0);
            s1 = wmma_f16(qf[kk], b1, s1);
        }
        // mask keys >= vl[b] (VALU select; EXEC stays full for WMMA)
        const bool ok0 = (kt + l16) < valid;
        const bool ok1 = (kt + 16 + l16) < valid;
#pragma unroll
        for (int r = 0; r < 8; ++r) {
            if (!ok0) s0[r] = -1e30f;
            if (!ok1) s1[r] = -1e30f;
        }

        // online softmax: row <-> (VGPR r, half-wave); 16-lane reductions
#pragma unroll
        for (int r = 0; r < 8; ++r) {
            const float a0 = s0[r], a1 = s1[r];
            float mx = fmaxf(a0, a1);
#pragma unroll
            for (int m = 8; m >= 1; m >>= 1) mx = fmaxf(mx, __shfl_xor(mx, m, 16));
            const float nm = fmaxf(mrun[r], mx);
            const float sc = __expf(mrun[r] - nm);
            const float p0 = __expf(a0 - nm);
            const float p1 = __expf(a1 - nm);
            float sum = p0 + p1;
#pragma unroll
            for (int m = 8; m >= 1; m >>= 1) sum += __shfl_xor(sum, m, 16);
            lrun[r] = lrun[r] * sc + sum;
            mrun[r] = nm;
#pragma unroll
            for (int dt = 0; dt < 8; ++dt) acc[dt][r] *= sc;
            // transpose P (C layout -> A layout) through per-wave LDS scratch
            _Float16* pr = &Ps[(w * 16 + (r + 8 * hf)) * PSTR];
            pr[l16]      = (_Float16)p0;
            pr[16 + l16] = (_Float16)p1;
        }
        // CDNA5 split counter: make per-wave LDS P stores visible before reload
        WAIT_DS0();

        v16h pf = make_frag(&Ps[(w * 16 + l16) * PSTR], kh);
#pragma unroll
        for (int dt = 0; dt < 8; ++dt) {
            v16h vf = make_frag(&Vs[(dt * 16 + l16) * VSTR], kh);
            acc[dt] = wmma_f16(pf, vf, acc[dt]);
        }
    }

    // write O[b, n, h*128 + d]
#pragma unroll
    for (int dt = 0; dt < 8; ++dt)
#pragma unroll
        for (int r = 0; r < 8; ++r) {
            const int row = nb0 + w * 16 + r + 8 * hf;
            const int d = dt * 16 + l16;
            const float ov = acc[dt][r] / fmaxf(lrun[r], 1e-20f);
            O[((size_t)b * SEQ + row) * ATTN_INNER + h * DIM_HEAD + d] = (_Float16)ov;
        }
}

// ---------------------------------------------------------------------------
// ffh = silu(gate) * x   (reads fused f16 tail, writes f16)
// ---------------------------------------------------------------------------
__global__ __launch_bounds__(256) void silu_gate_kernel(const _Float16* __restrict__ fused,
                                                        _Float16* __restrict__ ffh) {
    const size_t total = (size_t)ROWS * FF_INNER;
    size_t i = (size_t)blockIdx.x * blockDim.x + threadIdx.x;
    const size_t stride = (size_t)gridDim.x * blockDim.x;
    for (; i < total; i += stride) {
        const size_t row = i >> 13;           // / 8192
        const int j = (int)(i & 8191);
        const _Float16* fr = fused + row * FUSED_OUT + (ATTN_INNER + 2 * DIM_HEAD);
        const float xg = (float)fr[FF_INNER + j];   // gate = second half
        const float xx = (float)fr[j];              // x    = first half
        const float s = xg / (1.f + __expf(-xg));
        ffh[i] = (_Float16)(s * xx);
    }
}

// ---------------------------------------------------------------------------
// Host-side launcher
// ---------------------------------------------------------------------------
extern "C" void kernel_launch(void* const* d_in, const int* in_sizes, int n_in,
                              void* d_out, int out_size, void* d_ws, size_t ws_size,
                              hipStream_t stream) {
    const float* x      = (const float*)d_in[0];
    const int*   vl     = (const int*)d_in[1];
    const float* gamma  = (const float*)d_in[2];
    const float* wf32   = (const float*)d_in[3];
    const float* wa32   = (const float*)d_in[4];
    const float* wo32   = (const float*)d_in[5];
    float* out = (float*)d_out;

    // workspace carve-out (f16 buffers, 256B aligned)
    size_t off = 0;
    auto carve = [&](size_t bytes) {
        void* p = (char*)d_ws + off;
        off += (bytes + 255) & ~(size_t)255;
        return p;
    };
    _Float16* xn    = (_Float16*)carve((size_t)ROWS * DIM * 2);
    _Float16* wf16  = (_Float16*)carve((size_t)DIM * FUSED_OUT * 2);
    _Float16* wa16  = (_Float16*)carve((size_t)ATTN_INNER * DIM * 2);
    _Float16* wo16  = (_Float16*)carve((size_t)FF_INNER * DIM * 2);
    _Float16* fused = (_Float16*)carve((size_t)ROWS * FUSED_OUT * 2);
    _Float16* q16   = (_Float16*)carve((size_t)BATCH * HEADS * SEQ * DIM_HEAD * 2);
    _Float16* k16   = (_Float16*)carve((size_t)BATCH * SEQ * DIM_HEAD * 2);
    _Float16* v16   = (_Float16*)carve((size_t)BATCH * SEQ * DIM_HEAD * 2);
    _Float16* ao16  = (_Float16*)carve((size_t)ROWS * ATTN_INNER * 2);
    _Float16* ffh16 = (_Float16*)carve((size_t)ROWS * FF_INNER * 2);
    (void)ws_size; (void)n_in; (void)in_sizes; (void)out_size;

    // 1) LayerNorm -> f16
    ln_cast_kernel<<<ROWS, 256, 0, stream>>>(x, gamma, xn);

    // 2) weight casts
    cast_f32_f16_kernel<<<2048, 256, 0, stream>>>(wf32, wf16, (size_t)DIM * FUSED_OUT);
    cast_f32_f16_kernel<<<1024, 256, 0, stream>>>(wa32, wa16, (size_t)ATTN_INNER * DIM);
    cast_f32_f16_kernel<<<1024, 256, 0, stream>>>(wo32, wo16, (size_t)FF_INNER * DIM);

    // 3) fused = xn @ w_fused   (4096 x 18688 x 2048), f16 output
    gemm_wmma_kernel<true, false><<<dim3(FUSED_OUT / 128, ROWS / 128), 256, 0, stream>>>(
        xn, wf16, nullptr, fused, ROWS, FUSED_OUT, DIM);

    // 4) RoPE + split q/k/v
    rope_split_kernel<<<ROWS, 256, 0, stream>>>(fused, q16, k16, v16);

    // 5) attention
    attn_wmma_kernel<<<dim3(SEQ / 128, HEADS, BATCH), 256, 0, stream>>>(
        q16, k16, v16, vl, ao16);

    // 6) SwiGLU gate
    silu_gate_kernel<<<4096, 256, 0, stream>>>(fused, ffh16);

    // 7) out = ao @ w_attn_out ; out += ffh @ w_ff_out
    gemm_wmma_kernel<false, false><<<dim3(DIM / 128, ROWS / 128), 256, 0, stream>>>(
        ao16, wa16, out, nullptr, ROWS, DIM, ATTN_INNER);
    gemm_wmma_kernel<false, true><<<dim3(DIM / 128, ROWS / 128), 256, 0, stream>>>(
        ffh16, wo16, out, nullptr, ROWS, DIM, FF_INNER);
}
